// TransformerBlockMoE_5875515261168
// MI455X (gfx1250) — compile-verified, matrix operand order checked by reading, not run on hardware
//
#include <hip/hip_runtime.h>
#include <stdint.h>

// ---------------------------------------------------------------------------
// Shapes (compile-time constants from the reference)
// ---------------------------------------------------------------------------
namespace {
constexpr int BB   = 2;
constexpr int SS   = 1024;
constexpr int DD   = 768;
constexpr int HH   = 12;
constexpr int HDIM = 64;
constexpr int EE   = 8;
constexpr int DFF  = 3072;
constexpr int MTOK = BB * SS;      // 2048 tokens
}

// ---------------------------------------------------------------------------
// WMMA fragment types / helpers (CDNA5 wave32, v_wmma_f32_16x16x32_bf16)
// A-frag (16x32 bf16): lane l holds row m=l%16; K in {kb..kb+7, kb+16..kb+23},
// kb = (l>=16)?8:0, packed 2 elems/VGPR -> vector elems 0..7 = K kb..kb+7,
// elems 8..15 = K kb+16..kb+23.  B-frag mirrored (lane = column).
// ---------------------------------------------------------------------------
typedef __attribute__((ext_vector_type(16))) __bf16 v16bf;
typedef __attribute__((ext_vector_type(8)))  __bf16 v8bf;
typedef __attribute__((ext_vector_type(8)))  float  v8f;

__device__ __forceinline__ v16bf load_frag16(const __bf16* p) {
    v8bf lo = *(const v8bf*)(p);
    v8bf hi = *(const v8bf*)(p + 16);
    return __builtin_shufflevector(lo, hi, 0,1,2,3,4,5,6,7,8,9,10,11,12,13,14,15);
}

__device__ __forceinline__ v8f wmma_bf16(v16bf a, v16bf b, v8f c) {
    return __builtin_amdgcn_wmma_f32_16x16x32_bf16(
        /*neg_a=*/false, a, /*neg_b=*/false, b,
        /*c_mod=*/(short)0, c, /*reuse_a=*/false, /*reuse_b=*/false);
}

// ---------------------------------------------------------------------------
// LayerNorm: fp32 in -> bf16 out (feeds WMMA GEMMs)
// ---------------------------------------------------------------------------
__global__ __launch_bounds__(256) void ln_kernel(
    const float* __restrict__ x, const float* __restrict__ g,
    const float* __restrict__ b, __bf16* __restrict__ out)
{
    const int t = blockIdx.x;
    const int tid = threadIdx.x;
    const float* row = x + (size_t)t * DD;
    float s = 0.f, ss = 0.f;
    for (int i = tid; i < DD; i += 256) { float v = row[i]; s += v; ss += v * v; }
    __shared__ float sh1[256], sh2[256];
    sh1[tid] = s; sh2[tid] = ss;
    __syncthreads();
    for (int o = 128; o > 0; o >>= 1) {
        if (tid < o) { sh1[tid] += sh1[tid + o]; sh2[tid] += sh2[tid + o]; }
        __syncthreads();
    }
    const float mean = sh1[0] * (1.f / DD);
    const float var  = sh2[0] * (1.f / DD) - mean * mean;
    const float inv  = rsqrtf(var + 1e-5f);
    for (int i = tid; i < DD; i += 256)
        out[(size_t)t * DD + i] = (__bf16)((row[i] - mean) * inv * g[i] + b[i]);
}

// ---------------------------------------------------------------------------
// Generic tiled WMMA GEMM:  C[M,N] = act( A[M,K](bf16) x W[K,N](f32) + bias (+resid) )
// Block tile 128x128, 8 waves (4 over M x 2 over N), each wave 32x64
// (2 A-frags x 4 B-frags = 8 WMMAs per 32-K step).  W tile is converted
// fp32->bf16 into a double-buffered LDS [n][k] tile so the next tile's global
// loads overlap the current tile's WMMAs; one barrier per K-step.
// GATHER: A rows and C rows come from per-expert slot lists.
// ---------------------------------------------------------------------------
template<bool GATHER, bool RESID, bool GELU_EPI, bool OUTBF, int SHIFT>
__global__ __launch_bounds__(256) void gemm_wmma_kernel(
    const __bf16* __restrict__ A, int lda,
    const float* __restrict__ Wb, const float* __restrict__ biasb,
    const float* __restrict__ resid,
    const int* __restrict__ lists, const int* __restrict__ counts,
    float* __restrict__ Cf, __bf16* __restrict__ Cb,
    int N, int Kd)
{
    const int e = blockIdx.z;
    const float* W = Wb + (size_t)e * (size_t)Kd * (size_t)N;

    int cnt;
    const int* list = nullptr;
    if constexpr (GATHER) {
        cnt = counts[e];
        list = lists + (size_t)e * MTOK;
    } else {
        cnt = MTOK;
    }
    const int m0 = blockIdx.y * 128;
    if (m0 >= cnt) return;                       // uniform early-exit

    const int n_base = blockIdx.x * 128;
    const int tid  = threadIdx.x;
    const int lane = tid & 31;
    const int wave = tid >> 5;
    const int m_off = (wave & 3) * 32;
    const int n_off = (wave >> 2) * 64;
    const int nn = lane & 15;
    const int hi = lane >> 4;
    const int kb = hi << 3;                      // 0 or 8

    __shared__ __bf16 ldsB[2][128 * 32];         // [buf][n][k], 2 x 8 KB

    v8f acc0[4] = {}, acc1[4] = {};
    if (biasb != nullptr) {
        const float* bias = biasb + (size_t)e * N;
        for (int s = 0; s < 4; s++) {
            const float bv = bias[n_base + n_off + 16 * s + nn];
            for (int r = 0; r < 8; r++) { acc0[s][r] = bv; acc1[s][r] = bv; }
        }
    }

    // Resolve this lane's two A rows once (K-contiguous afterwards)
    int arow0, arow1;
    {
        const int pa0 = m0 + m_off + nn;
        const int pa1 = pa0 + 16;
        if constexpr (GATHER) {
            const int p0 = pa0 < cnt ? pa0 : cnt - 1;
            const int p1 = pa1 < cnt ? pa1 : cnt - 1;
            arow0 = list[p0] >> SHIFT;
            arow1 = list[p1] >> SHIFT;
        } else {
            arow0 = pa0;
            arow1 = pa1;
        }
    }
    const __bf16* Arow0 = A + (size_t)arow0 * lda;
    const __bf16* Arow1 = A + (size_t)arow1 * lda;

    // Per-thread slice of the cooperative W-tile copy (32 k x 128 n fp32)
    const int kk = tid >> 3;               // 0..31 : k row
    const int nf = (tid & 7) << 4;         // 0..112 : 16-float column chunk
    float4 wreg[4];

    auto load_tile = [&](int k0) {
        const float* src = W + (size_t)(k0 + kk) * N + n_base + nf;
        wreg[0] = *(const float4*)(src + 0);
        wreg[1] = *(const float4*)(src + 4);
        wreg[2] = *(const float4*)(src + 8);
        wreg[3] = *(const float4*)(src + 12);
    };
    auto store_tile = [&](__bf16* dst) {
        for (int i = 0; i < 4; i++) {
            const int n = nf + i * 4;
            dst[(n + 0) * 32 + kk] = (__bf16)wreg[i].x;
            dst[(n + 1) * 32 + kk] = (__bf16)wreg[i].y;
            dst[(n + 2) * 32 + kk] = (__bf16)wreg[i].z;
            dst[(n + 3) * 32 + kk] = (__bf16)wreg[i].w;
        }
    };

    const int NK = Kd / 32;
    load_tile(0);
    store_tile(ldsB[0]);
    __syncthreads();

    int cur = 0;
    for (int t = 0; t < NK; ++t) {
        const int k0 = t * 32;
        if (t + 1 < NK) load_tile(k0 + 32);          // overlap with WMMAs below
        if (t + 2 < NK)                               // warm L2 one tile further
            __builtin_prefetch(W + (size_t)(k0 + 64 + kk) * N + n_base + nf, 0, 1);

        const v16bf a0 = load_frag16(Arow0 + k0 + kb);
        const v16bf a1 = load_frag16(Arow1 + k0 + kb);
        const __bf16* bbase = ldsB[cur];
        for (int s = 0; s < 4; s++) {
            const v16bf bf = load_frag16(&bbase[(n_off + 16 * s + nn) * 32 + kb]);
            acc0[s] = wmma_bf16(a0, bf, acc0[s]);
            acc1[s] = wmma_bf16(a1, bf, acc1[s]);
        }

        if (t + 1 < NK) store_tile(ldsB[cur ^ 1]);
        __syncthreads();
        cur ^= 1;
    }

    // Epilogue: C layout -> lane holds col nn, rows r+8*hi (r = vector elem)
    for (int f = 0; f < 2; f++) {
        v8f* accf = (f == 0) ? acc0 : acc1;
        for (int s = 0; s < 4; s++) {
            const int col = n_base + n_off + 16 * s + nn;
            for (int r = 0; r < 8; r++) {
                const int p = m0 + m_off + f * 16 + r + 8 * hi;
                if constexpr (GATHER) { if (p >= cnt) continue; }
                int crow;
                if constexpr (GATHER) crow = list[p]; else crow = p;
                float v = accf[s][r];
                if constexpr (RESID)    v += resid[(size_t)crow * N + col];
                if constexpr (GELU_EPI) v = 0.5f * v * (1.f + erff(v * 0.70710678118654752f));
                if constexpr (OUTBF) Cb[(size_t)crow * N + col] = (__bf16)v;
                else                 Cf[(size_t)crow * N + col] = v;
            }
        }
    }
}

// ---------------------------------------------------------------------------
// Flash-style causal attention. 128 threads = 4 waves, 64 q rows / block,
// one wave owns 16 q rows. 32-key chunks staged in LDS; online softmax via
// __shfl_xor across the 16-lane column groups of the WMMA C layout.
// ---------------------------------------------------------------------------
__global__ __launch_bounds__(128) void attn_kernel(
    const __bf16* __restrict__ Q, const __bf16* __restrict__ Kb,
    const __bf16* __restrict__ Vb, __bf16* __restrict__ O)
{
    const int q0  = blockIdx.x * 64;
    const int h   = blockIdx.y;
    const int b   = blockIdx.z;
    const int tid = threadIdx.x;
    const int lane = tid & 31;
    const int wave = tid >> 5;
    const int nn = lane & 15;
    const int hi = lane >> 4;
    const int kb = hi << 3;
    const int qm0 = q0 + wave * 16;

    __shared__ __bf16 ldsK[32 * 64];      // [key][d]
    __shared__ __bf16 ldsV[64 * 32];      // [d][key] (transposed)
    __shared__ __bf16 ldsP[4][16 * 32];   // per-wave P staging [q][key]

    // Q fragments for this wave (row = qm0+nn), d split into two K=32 frags
    const __bf16* qp = Q + ((size_t)(b * SS + qm0 + nn) * DD + h * HDIM + kb);
    const v16bf aq0 = load_frag16(qp);
    const v16bf aq1 = load_frag16(qp + 32);

    v8f o0 = {}, o1 = {}, o2 = {}, o3 = {};
    float mrow[8], lrow[8];
    for (int r = 0; r < 8; r++) { mrow[r] = -INFINITY; lrow[r] = 0.f; }

    for (int j0 = 0; j0 < q0 + 64; j0 += 32) {
        // Cooperative K/V chunk load (32 keys x 64 d)
        for (int i = 0; i < 2; i++) {
            const int li   = tid + i * 128;
            const int key  = li >> 3;
            const int doff = (li & 7) * 8;
            const size_t src = (size_t)(b * SS + j0 + key) * DD + h * HDIM + doff;
            *(v8bf*)&ldsK[key * 64 + doff] = *(const v8bf*)(Kb + src);
            const v8bf vv = *(const v8bf*)(Vb + src);
            for (int j = 0; j < 8; j++) ldsV[(doff + j) * 32 + key] = vv[j];
        }
        __syncthreads();

        // Scores: two 16x16 tiles (keys j0..j0+15, j0+16..j0+31)
        v8f sacc[2];
        for (int nt = 0; nt < 2; nt++) {
            const int key = nt * 16 + nn;
            v8f z = {};
            z = wmma_bf16(aq0, load_frag16(&ldsK[key * 64 + kb]),      z);
            z = wmma_bf16(aq1, load_frag16(&ldsK[key * 64 + 32 + kb]), z);
            sacc[nt] = z;
        }

        // Scale + causal mask + online softmax
        float p0[8], p1[8], cmax[8];
        for (int r = 0; r < 8; r++) {
            const int qrow = qm0 + r + 8 * hi;
            float s0 = sacc[0][r] * 0.125f;
            float s1 = sacc[1][r] * 0.125f;
            if (j0 + nn      > qrow) s0 = -1e30f;
            if (j0 + 16 + nn > qrow) s1 = -1e30f;
            p0[r] = s0; p1[r] = s1;
            cmax[r] = fmaxf(s0, s1);
        }
        for (int r = 0; r < 8; r++)
            for (int o = 1; o < 16; o <<= 1)
                cmax[r] = fmaxf(cmax[r], __shfl_xor(cmax[r], o, 32));

        float alpha[8], rs[8];
        for (int r = 0; r < 8; r++) {
            const float mn = fmaxf(mrow[r], cmax[r]);
            alpha[r] = expf(mrow[r] - mn);
            mrow[r]  = mn;
            p0[r] = expf(p0[r] - mn);
            p1[r] = expf(p1[r] - mn);
            rs[r] = p0[r] + p1[r];
        }
        for (int r = 0; r < 8; r++)
            for (int o = 1; o < 16; o <<= 1)
                rs[r] += __shfl_xor(rs[r], o, 32);
        for (int r = 0; r < 8; r++) {
            lrow[r] = lrow[r] * alpha[r] + rs[r];
            o0[r] *= alpha[r]; o1[r] *= alpha[r];
            o2[r] *= alpha[r]; o3[r] *= alpha[r];
        }

        // C layout -> A layout via per-wave LDS (same-wave DS ops are in-order)
        __bf16* pw = ldsP[wave];
        for (int r = 0; r < 8; r++) {
            const int q = r + 8 * hi;
            pw[q * 32 + nn]      = (__bf16)p0[r];
            pw[q * 32 + 16 + nn] = (__bf16)p1[r];
        }
        const v16bf ap = load_frag16(&pw[nn * 32 + kb]);

        // PV: B element (k=key, n=d) = ldsV[d][key]
        o0 = wmma_bf16(ap, load_frag16(&ldsV[(0  + nn) * 32 + kb]), o0);
        o1 = wmma_bf16(ap, load_frag16(&ldsV[(16 + nn) * 32 + kb]), o1);
        o2 = wmma_bf16(ap, load_frag16(&ldsV[(32 + nn) * 32 + kb]), o2);
        o3 = wmma_bf16(ap, load_frag16(&ldsV[(48 + nn) * 32 + kb]), o3);
        __syncthreads();
    }

    for (int r = 0; r < 8; r++) {
        const size_t row = (size_t)(b * SS + qm0 + r + 8 * hi) * DD + h * HDIM;
        const float inv = 1.f / lrow[r];
        O[row + 0  + nn] = (__bf16)(o0[r] * inv);
        O[row + 16 + nn] = (__bf16)(o1[r] * inv);
        O[row + 32 + nn] = (__bf16)(o2[r] * inv);
        O[row + 48 + nn] = (__bf16)(o3[r] * inv);
    }
}

// ---------------------------------------------------------------------------
// Router: logits = xn2 @ Wr + br   (E=8, tiny; one 32-lane group per token)
// ---------------------------------------------------------------------------
__global__ __launch_bounds__(256) void router_kernel(
    const __bf16* __restrict__ xn, const float* __restrict__ Wr,
    const float* __restrict__ br, float* __restrict__ logits)
{
    const int tid = threadIdx.x;
    const int lane = tid & 31;
    const int t = blockIdx.x * 8 + (tid >> 5);
    const __bf16* row = xn + (size_t)t * DD;
    float acc[EE] = {};
    for (int d = lane; d < DD; d += 32) {
        const float xv = (float)row[d];
        const float* w = Wr + (size_t)d * EE;
        for (int e = 0; e < EE; e++) acc[e] += xv * w[e];
    }
    for (int e = 0; e < EE; e++)
        for (int o = 1; o < 32; o <<= 1)
            acc[e] += __shfl_xor(acc[e], o, 32);
    if (lane == 0)
        for (int e = 0; e < EE; e++)
            logits[(size_t)t * EE + e] = acc[e] + br[e];
}

// ---------------------------------------------------------------------------
// Top-2 + gates + routing lists (slot = 2*token + k; every slot written once
// so per-slot GEMM results are independent of atomic ordering -> deterministic)
// ---------------------------------------------------------------------------
__global__ __launch_bounds__(256) void topk_kernel(
    const float* __restrict__ logits, float* __restrict__ gates,
    int* __restrict__ counts, int* __restrict__ lists)
{
    const int t = blockIdx.x * blockDim.x + threadIdx.x;
    if (t >= MTOK) return;
    const float* l = logits + (size_t)t * EE;
    int i0 = 0; float v0 = l[0];
    for (int e = 1; e < EE; e++) if (l[e] > v0) { v0 = l[e]; i0 = e; }
    int i1 = (i0 == 0) ? 1 : 0; float v1 = l[i1];
    for (int e = 0; e < EE; e++) if (e != i0 && l[e] > v1) { v1 = l[e]; i1 = e; }
    const float e1 = expf(v1 - v0);
    const float den = 1.f + e1;
    gates[2 * t]     = 1.f / den;
    gates[2 * t + 1] = e1 / den;
    const int p0 = atomicAdd(&counts[i0], 1);
    lists[(size_t)i0 * MTOK + p0] = 2 * t;
    const int p1 = atomicAdd(&counts[i1], 1);
    lists[(size_t)i1 * MTOK + p1] = 2 * t + 1;
}

// ---------------------------------------------------------------------------
// Combine: out = hres + g0 * eo[slot0] + g1 * eo[slot1]
// ---------------------------------------------------------------------------
__global__ __launch_bounds__(256) void combine_kernel(
    const float* __restrict__ hres, const float* __restrict__ eo,
    const float* __restrict__ gates, float* __restrict__ out)
{
    const int i = blockIdx.x * 256 + threadIdx.x;
    if (i >= MTOK * DD) return;
    const int t = i / DD;
    const int d = i - t * DD;
    out[i] = hres[i]
           + gates[2 * t]     * eo[(size_t)(2 * t)     * DD + d]
           + gates[2 * t + 1] * eo[(size_t)(2 * t + 1) * DD + d];
}

// ---------------------------------------------------------------------------
// Host orchestration
// ---------------------------------------------------------------------------
extern "C" void kernel_launch(void* const* d_in, const int* in_sizes, int n_in,
                              void* d_out, int out_size, void* d_ws, size_t ws_size,
                              hipStream_t stream)
{
    (void)in_sizes; (void)n_in; (void)out_size; (void)ws_size;
    const float* x     = (const float*)d_in[0];
    const float* ln1_g = (const float*)d_in[1];
    const float* ln1_b = (const float*)d_in[2];
    const float* Wq    = (const float*)d_in[3];
    const float* Wk    = (const float*)d_in[4];
    const float* Wv    = (const float*)d_in[5];
    const float* Wo    = (const float*)d_in[6];
    const float* bo    = (const float*)d_in[7];
    const float* ln2_g = (const float*)d_in[8];
    const float* ln2_b = (const float*)d_in[9];
    const float* Wr    = (const float*)d_in[10];
    const float* br    = (const float*)d_in[11];
    const float* W1    = (const float*)d_in[12];
    const float* b1    = (const float*)d_in[13];
    const float* W2    = (const float*)d_in[14];
    const float* b2    = (const float*)d_in[15];
    float* out = (float*)d_out;

    // Workspace carve-out (~63 MB)
    uint8_t* w = (uint8_t*)d_ws;
    auto alloc = [&](size_t bytes) -> uint8_t* {
        uint8_t* p = w;
        w += (bytes + 255) & ~(size_t)255;
        return p;
    };
    __bf16* xn1   = (__bf16*)alloc((size_t)MTOK * DD * 2);
    __bf16* qb    = (__bf16*)alloc((size_t)MTOK * DD * 2);
    __bf16* kbuf  = (__bf16*)alloc((size_t)MTOK * DD * 2);
    __bf16* vbuf  = (__bf16*)alloc((size_t)MTOK * DD * 2);
    __bf16* ao    = (__bf16*)alloc((size_t)MTOK * DD * 2);
    float*  hres  = (float*)alloc((size_t)MTOK * DD * 4);
    __bf16* xn2   = (__bf16*)alloc((size_t)MTOK * DD * 2);
    float*  logit = (float*)alloc((size_t)MTOK * EE * 4);
    float*  gates = (float*)alloc((size_t)MTOK * 2 * 4);
    int*    cnts  = (int*)alloc(EE * 4);
    int*    lists = (int*)alloc((size_t)EE * MTOK * 4);
    __bf16* hmoe  = (__bf16*)alloc((size_t)2 * MTOK * DFF * 2);
    float*  eo    = (float*)alloc((size_t)2 * MTOK * DD * 4);

    hipMemsetAsync(cnts, 0, EE * sizeof(int), stream);

    // ---- LN1 -> xn1 (bf16) ----
    ln_kernel<<<MTOK, 256, 0, stream>>>(x, ln1_g, ln1_b, xn1);

    // ---- QKV projections (bf16 WMMA, no bias) ----
    dim3 gP(DD / 128, MTOK / 128, 1);
    gemm_wmma_kernel<false,false,false,true,0><<<gP, 256, 0, stream>>>(
        xn1, DD, Wq, nullptr, nullptr, nullptr, nullptr, nullptr, qb,   DD, DD);
    gemm_wmma_kernel<false,false,false,true,0><<<gP, 256, 0, stream>>>(
        xn1, DD, Wk, nullptr, nullptr, nullptr, nullptr, nullptr, kbuf, DD, DD);
    gemm_wmma_kernel<false,false,false,true,0><<<gP, 256, 0, stream>>>(
        xn1, DD, Wv, nullptr, nullptr, nullptr, nullptr, nullptr, vbuf, DD, DD);

    // ---- causal flash attention ----
    attn_kernel<<<dim3(SS / 64, HH, BB), 128, 0, stream>>>(qb, kbuf, vbuf, ao);

    // ---- Wo projection + bias + residual -> hres (fp32) ----
    gemm_wmma_kernel<false,true,false,false,0><<<gP, 256, 0, stream>>>(
        ao, DD, Wo, bo, x, nullptr, nullptr, hres, nullptr, DD, DD);

    // ---- LN2 -> xn2 (bf16) ----
    ln_kernel<<<MTOK, 256, 0, stream>>>(hres, ln2_g, ln2_b, xn2);

    // ---- router + top-2 routing ----
    router_kernel<<<MTOK / 8, 256, 0, stream>>>(xn2, Wr, br, logit);
    topk_kernel<<<MTOK / 256, 256, 0, stream>>>(logit, gates, cnts, lists);

    // ---- expert GEMM 1: gathered tokens, GELU epilogue, bf16 out ----
    gemm_wmma_kernel<true,false,true,true,1><<<dim3(DFF / 128, MTOK / 128, EE), 256, 0, stream>>>(
        xn2, DD, W1, b1, nullptr, lists, cnts, nullptr, hmoe, DFF, DD);

    // ---- expert GEMM 2: gathered hidden rows, bias, fp32 out ----
    gemm_wmma_kernel<true,false,false,false,0><<<dim3(DD / 128, MTOK / 128, EE), 256, 0, stream>>>(
        hmoe, DFF, W2, b2, nullptr, lists, cnts, eo, nullptr, DD, DFF);

    // ---- gate-weighted combine + residual ----
    combine_kernel<<<(MTOK * DD) / 256, 256, 0, stream>>>(hres, eo, gates, out);
}
